// PositionAttentionModule_23192823398747
// MI455X (gfx1250) — matrix operand order repaired and checked by reading, not compiled, not run on hardware
//
#include <hip/hip_runtime.h>
#include <hip/hip_bf16.h>

// PositionAttentionModule for MI455X (gfx1250), bf16 WMMA pipeline.
// B=4, C=256, C8=32, N=4096. Dominant GEMM: f3(256x4096) @ attn(4096x4096).
//
// All WMMA B operands live in "packed-B" layout = the
// v_wmma_f32_16x16x32_bf16 B-fragment image in this code's contiguous k order (k = 16*hi + s):
//   tile (kb = k>>5, nt = m>>4) of 512 bf16; element (k,m) lives at
//   tile*512 + ((m&15) + 16*((k&31)>=16))*16 + (k&15)
// so a B fragment is two aligned 16B loads per lane, and a 32K x 128N
// panel is one contiguous 8KB region (ideal for LDS staging).

typedef __bf16 bf16_t;
typedef __attribute__((ext_vector_type(16))) __bf16 v16bf;
typedef __attribute__((ext_vector_type(8)))  __bf16 v8bf;
typedef __attribute__((ext_vector_type(8)))  float   v8f;

union FragAB {
    v16bf  v;
    v8bf   h[2];
    __bf16 e[16];
};

// ---------------------------------------------------------------- converts
__global__ void cvt_f32_bf16(const float* __restrict__ in,
                             bf16_t* __restrict__ out, size_t n) {
    size_t i = (size_t)blockIdx.x * blockDim.x + threadIdx.x;
    if (i < n) out[i] = (bf16_t)in[i];
}

// x [B,C,N] fp32 -> packed-B bf16 (batch stride C*N elements)
__global__ void pack_x_bf16(const float* __restrict__ in,
                            bf16_t* __restrict__ out, int C_, int N_, size_t total) {
    size_t i = (size_t)blockIdx.x * blockDim.x + threadIdx.x;
    if (i >= total) return;
    const int    n = (int)(i % N_);
    const int    c = (int)((i / N_) % C_);
    const size_t b = i / ((size_t)N_ * C_);
    const size_t o = b * (size_t)C_ * N_
                   + (size_t)((c >> 5) * (N_ >> 4) + (n >> 4)) * 512
                   + (size_t)((n & 15) + (((c & 31) >= 16) ? 16 : 0)) * 16
                   + (c & 15);
    out[o] = (bf16_t)in[i];
}

// ---------------------------------------------------------------- conv GEMM
// out[M,N] = A[M,K] (row-major bf16) x B[K,N] (packed-B bf16)
// Each wave owns 16 x 64 (4 adjacent packed N-tiles): A fragment loaded once
// per K-step, 4 independent WMMAs issued back-to-back.
// OLAYOUT 0: bf16 row-major (f3) | 1: packed-B (f2) | 2: transposed [NxM] (f1)
// OLAYOUT 3: f32 row-major (final conv)
template <int OLAYOUT>
__global__ __launch_bounds__(256)
void gemm_bf16_wmma(const bf16_t* __restrict__ A,
                    const bf16_t* __restrict__ Bm,
                    void* __restrict__ outp,
                    const float* __restrict__ bias,
                    int M, int N, int K,
                    size_t strideB, size_t strideOut)
{
    const int lane   = threadIdx.x & 31;
    const int wave   = threadIdx.x >> 5;
    const int ntile0 = (blockIdx.x * 8 + wave) * 4;
    const int mtile  = blockIdx.y;
    const int b      = blockIdx.z;
    if (ntile0 * 16 >= N || mtile * 16 >= M) return;

    const bf16_t* Bb = Bm + strideB * b;

    const int rloc   = lane & 15;
    const int hi     = (lane >> 4) & 1;
    const int arow   = mtile * 16 + rloc;
    const int koffA  = hi ? 16 : 0;
    const int ktiles = N >> 4;

    const bf16_t* bbase = Bb + (size_t)ntile0 * 512 + (size_t)lane * 16;
    const size_t  brow  = (size_t)ktiles * 512;     // one 32-K block

    v8f acc[4] = {};
#pragma unroll 2
    for (int kk = 0; kk < K; kk += 32) {
        FragAB a;
        const bf16_t* ap = A + (size_t)arow * K + kk + koffA;
        a.h[0] = *(const v8bf*)(ap);
        a.h[1] = *(const v8bf*)(ap + 8);

        const bf16_t* bp = bbase + (size_t)(kk >> 5) * brow;
        if (kk + 32 < K)
            __builtin_prefetch(bp + brow, 0, 1);

#pragma unroll
        for (int j = 0; j < 4; ++j) {
            FragAB bf;
            bf.h[0] = *(const v8bf*)(bp + j * 512);
            bf.h[1] = *(const v8bf*)(bp + j * 512 + 8);
            acc[j] = __builtin_amdgcn_wmma_f32_16x16x32_bf16(
                false, a.v, false, bf.v, (short)0, acc[j], false, false);
        }
    }

    // C/D layout: VGPR r -> M = r + 8*hi, N = lane&15
#pragma unroll
    for (int j = 0; j < 4; ++j) {
        const int ntile = ntile0 + j;
        const int bcol  = ntile * 16 + rloc;
        if (OLAYOUT == 1 || OLAYOUT == 2) {
            v8bf ov;
#pragma unroll
            for (int r = 0; r < 8; ++r) {
                const int orow = mtile * 16 + r + (hi ? 8 : 0);
                float v = acc[j][r];
                if (bias) v += bias[orow];
                ov[r] = (bf16_t)v;
            }
            bf16_t* ob;
            if (OLAYOUT == 2) {
                ob = (bf16_t*)outp + strideOut * b
                   + (size_t)bcol * M + mtile * 16 + 8 * hi;
            } else {
                ob = (bf16_t*)outp + strideOut * b
                   + (size_t)((mtile >> 1) * ktiles + ntile) * 512
                   + (size_t)(rloc + ((mtile & 1) ? 16 : 0)) * 16
                   + 8 * hi;
            }
            *(v8bf*)ob = ov;
        } else {
#pragma unroll
            for (int r = 0; r < 8; ++r) {
                const int    orow = mtile * 16 + r + (hi ? 8 : 0);
                const size_t oidx = strideOut * b + (size_t)orow * N + bcol;
                float v = acc[j][r];
                if (bias) v += bias[orow];
                if (OLAYOUT == 3) ((float*)outp)[oidx] = v;
                else              ((bf16_t*)outp)[oidx] = (bf16_t)v;
            }
        }
    }
}

// ---------------------------------------------------------------- big GEMM
// gated = (f3 @ attn) * x, LDS-staged & double-buffered.
// Block = 256 threads (8 waves), block tile 64(M) x 128(N), K-steps of 32.
// Per step: A panel 64x32 (4KB) + B panel 32x128 packed (8KB, contiguous!)
// copied to LDS; each wave computes 32x32 (4 independent WMMAs) from LDS.
// Arithmetic intensity ~43 flops/byte of global traffic -> matrix-pipe bound.
#define BM 64
#define BN 128
__global__ __launch_bounds__(256)
void attn_apply_wmma(const bf16_t* __restrict__ A,    // f3 row-major [M x K]
                     const bf16_t* __restrict__ Bp,   // attn packed-B [K x N]
                     const float* __restrict__ gate,  // x row-major f32 [M x N]
                     bf16_t* __restrict__ outp,       // gated, packed-B
                     int M, int N, int K)
{
    __shared__ bf16_t ldsA[2][BM * 32];       // 2 x 4KB
    __shared__ bf16_t ldsB[2][32 * BN];       // 2 x 8KB (8 packed tiles)

    const int tid  = threadIdx.x;
    const int lane = tid & 31;
    const int wave = tid >> 5;
    const int wm   = wave >> 2;               // 0..1 : wave row (32 rows each)
    const int wn   = wave & 3;                // 0..3 : wave col (32 cols each)
    const int m0   = blockIdx.y * BM;
    const int n0   = blockIdx.x * BN;
    const int ktiles = N >> 4;

    const int rloc  = lane & 15;
    const int hi    = (lane >> 4) & 1;
    const int koffA = hi ? 16 : 0;

    // staging assignment: A 4KB -> 16B/thread, B 8KB -> 32B/thread
    const int    arow_cp = tid >> 2;                  // 0..63
    const int    aseg    = (tid & 3) * 8;             // 0,8,16,24
    const size_t bblk    = (size_t)ktiles * 512;      // B advance per 32-K
    const bf16_t* bpanel = Bp + (size_t)(n0 >> 4) * 512 + (size_t)tid * 16;
    const bf16_t* apanel = A + (size_t)(m0 + arow_cp) * K + aseg;

    v8bf ra, rb0, rb1;
    // prologue: fetch k-block 0 and park it in LDS buffer 0
    ra  = *(const v8bf*)(apanel);
    rb0 = *(const v8bf*)(bpanel);
    rb1 = *(const v8bf*)(bpanel + 4096);
    *(v8bf*)(&ldsA[0][arow_cp * 32 + aseg]) = ra;
    *(v8bf*)(&ldsB[0][tid * 16])        = rb0;
    *(v8bf*)(&ldsB[0][tid * 16 + 4096]) = rb1;

    v8f acc[2][2] = {};
    const int nsteps = K >> 5;
    for (int i = 0; i < nsteps; ++i) {
        __syncthreads();
        const int cur = i & 1;
        const bool more = (i + 1) < nsteps;
        if (more) {
            const int kk = (i + 1) << 5;
            ra  = *(const v8bf*)(apanel + kk);
            rb0 = *(const v8bf*)(bpanel + (size_t)(i + 1) * bblk);
            rb1 = *(const v8bf*)(bpanel + (size_t)(i + 1) * bblk + 4096);
        }

        // compute 32x32 per wave from LDS: 2 A frags, 2 B frags, 4 WMMAs
        FragAB a[2];
#pragma unroll
        for (int im = 0; im < 2; ++im) {
            const bf16_t* ap = &ldsA[cur][(wm * 32 + im * 16 + rloc) * 32 + koffA];
            a[im].h[0] = *(const v8bf*)(ap);
            a[im].h[1] = *(const v8bf*)(ap + 8);
        }
#pragma unroll
        for (int jn = 0; jn < 2; ++jn) {
            const bf16_t* bp = &ldsB[cur][(wn * 2 + jn) * 512 + lane * 16];
            FragAB bf;
            bf.h[0] = *(const v8bf*)(bp);
            bf.h[1] = *(const v8bf*)(bp + 8);
#pragma unroll
            for (int im = 0; im < 2; ++im)
                acc[im][jn] = __builtin_amdgcn_wmma_f32_16x16x32_bf16(
                    false, a[im].v, false, bf.v, (short)0, acc[im][jn], false, false);
        }

        if (more) {
            __syncthreads();
            const int nxt = 1 - cur;
            *(v8bf*)(&ldsA[nxt][arow_cp * 32 + aseg]) = ra;
            *(v8bf*)(&ldsB[nxt][tid * 16])        = rb0;
            *(v8bf*)(&ldsB[nxt][tid * 16 + 4096]) = rb1;
        }
    }

    // epilogue: gate by x, write packed-B (one b128 store per tile)
#pragma unroll
    for (int im = 0; im < 2; ++im) {
#pragma unroll
        for (int jn = 0; jn < 2; ++jn) {
            const int mtile = (m0 >> 4) + wm * 2 + im;
            const int ntile = (n0 >> 4) + wn * 2 + jn;
            const int bcol  = ntile * 16 + rloc;
            v8bf ov;
#pragma unroll
            for (int r = 0; r < 8; ++r) {
                const int orow = mtile * 16 + r + (hi ? 8 : 0);
                ov[r] = (bf16_t)(acc[im][jn][r] * gate[(size_t)orow * N + bcol]);
            }
            bf16_t* ob = outp
                + (size_t)((mtile >> 1) * ktiles + ntile) * 512
                + (size_t)(rloc + ((mtile & 1) ? 16 : 0)) * 16
                + 8 * hi;
            *(v8bf*)ob = ov;
        }
    }
}

// ---------------------------------------------------------------- fused logits + softmax
// attn[n0:n0+16, :] = softmax_over_m( f1t(16x32) @ f2(32x4096) ), written
// packed-B. 16x4096 fp32 score strip in LDS (256KB of 320KB/WGP).
__global__ __launch_bounds__(256)
void attn_logits_softmax(const bf16_t* __restrict__ f1t,
                         const bf16_t* __restrict__ f2p,
                         bf16_t* __restrict__ attnp, int N)
{
    extern __shared__ char smem_raw[];
    float* srow  = (float*)smem_raw;         // [16][N]
    float* part  = srow + (size_t)16 * N;    // [16][16]
    float* rstat = part + 256;               // [16]

    const int lane = threadIdx.x & 31;
    const int wave = threadIdx.x >> 5;
    const int rloc = lane & 15;
    const int hi   = (lane >> 4) & 1;
    const int n0   = blockIdx.x * 16;

    FragAB a;
    {
        const bf16_t* ap = f1t + (size_t)(n0 + rloc) * 32 + (hi ? 16 : 0);
        a.h[0] = *(const v8bf*)(ap);
        a.h[1] = *(const v8bf*)(ap + 8);
    }

    for (int mt = wave; mt < (N >> 4); mt += 8) {
        FragAB bf;
        const bf16_t* bp = f2p + (size_t)mt * 512 + (size_t)lane * 16;
        bf.h[0] = *(const v8bf*)(bp);
        bf.h[1] = *(const v8bf*)(bp + 8);
        v8f s = {};
        s = __builtin_amdgcn_wmma_f32_16x16x32_bf16(
            false, a.v, false, bf.v, (short)0, s, false, false);
        const int col = mt * 16 + rloc;
#pragma unroll
        for (int r = 0; r < 8; ++r)
            srow[(size_t)(r + (hi ? 8 : 0)) * N + col] = s[r];
    }
    __syncthreads();

    const int row   = threadIdx.x & 15;
    const int chunk = threadIdx.x >> 4;
    const int cw    = N >> 4;
    float* rowp = srow + (size_t)row * N + chunk * cw;

    float mx = -3.0e38f;
    for (int i = 0; i < cw; ++i) mx = fmaxf(mx, rowp[i]);
    part[row * 16 + chunk] = mx;
    __syncthreads();
    if (threadIdx.x < 16) {
        float m = part[threadIdx.x * 16];
        for (int i = 1; i < 16; ++i) m = fmaxf(m, part[threadIdx.x * 16 + i]);
        rstat[threadIdx.x] = m;
    }
    __syncthreads();
    const float rm = rstat[row];
    float sum = 0.f;
    for (int i = 0; i < cw; ++i) {
        float e = __expf(rowp[i] - rm);
        rowp[i] = e;
        sum += e;
    }
    part[row * 16 + chunk] = sum;
    __syncthreads();
    if (threadIdx.x < 16) {
        float s = 0.f;
        for (int i = 0; i < 16; ++i) s += part[threadIdx.x * 16 + i];
        rstat[threadIdx.x] = 1.0f / s;
    }
    __syncthreads();
    const float inv = rstat[row];

    const int    k     = n0 + row;               // K index in big GEMM's B
    const int    i_e   = k & 15;
    const int    hioff = ((k & 31) >= 16) ? 16 : 0;
    bf16_t* kbase = attnp + (size_t)(k >> 5) * (N >> 4) * 512;
    for (int c = 0; c < cw; ++c) {
        const int col = chunk * cw + c;
        kbase[(size_t)(col >> 4) * 512 + (size_t)((col & 15) + hioff) * 16 + i_e]
            = (bf16_t)(rowp[c] * inv);
    }
}

// ---------------------------------------------------------------- launch
extern "C" void kernel_launch(void* const* d_in, const int* in_sizes, int n_in,
                              void* d_out, int out_size, void* d_ws, size_t ws_size,
                              hipStream_t stream)
{
    (void)in_sizes; (void)n_in; (void)out_size; (void)ws_size;
    const int Bsz = 4, C = 256, C8 = 32, N = 4096;

    const float* x  = (const float*)d_in[0];
    const float* W1 = (const float*)d_in[1];
    const float* b1 = (const float*)d_in[2];
    const float* W2 = (const float*)d_in[3];
    const float* b2 = (const float*)d_in[4];
    const float* W3 = (const float*)d_in[5];
    const float* b3 = (const float*)d_in[6];
    const float* Wo = (const float*)d_in[7];
    const float* bo = (const float*)d_in[8];
    float* out = (float*)d_out;

    char*  ws  = (char*)d_ws;
    size_t off = 0;
    auto take = [&](size_t bytes) -> char* {
        char* p = ws + off;
        off = (off + bytes + 255) & ~(size_t)255;
        return p;
    };
    bf16_t* xbp  = (bf16_t*)take((size_t)Bsz * C * N * 2);   // x, packed-B
    bf16_t* w1b  = (bf16_t*)take((size_t)C8 * C * 2);
    bf16_t* w2b  = (bf16_t*)take((size_t)C8 * C * 2);
    bf16_t* w3b  = (bf16_t*)take((size_t)C * C * 2);
    bf16_t* wob  = (bf16_t*)take((size_t)C * C * 2);
    bf16_t* f1t  = (bf16_t*)take((size_t)Bsz * N * C8 * 2);  // transposed [N x 32]
    bf16_t* f2p  = (bf16_t*)take((size_t)Bsz * C8 * N * 2);  // packed-B
    bf16_t* f3b  = (bf16_t*)take((size_t)Bsz * C * N * 2);   // row-major (A operand)
    bf16_t* attn = (bf16_t*)take((size_t)N * N * 2);         // packed-B, reused per batch
    bf16_t* gbp  = (bf16_t*)take((size_t)Bsz * C * N * 2);   // gated, packed-B

    auto cvt = [&](const float* in, bf16_t* o, size_t n) {
        cvt_f32_bf16<<<dim3((unsigned)((n + 255) / 256)), dim3(256), 0, stream>>>(in, o, n);
    };
    const size_t xtot = (size_t)Bsz * C * N;
    pack_x_bf16<<<dim3((unsigned)((xtot + 255) / 256)), dim3(256), 0, stream>>>(
        x, xbp, C, N, xtot);
    cvt(W1, w1b, (size_t)C8 * C);
    cvt(W2, w2b, (size_t)C8 * C);
    cvt(W3, w3b, (size_t)C * C);
    cvt(Wo, wob, (size_t)C * C);

    const dim3 blk(256);
    // f1 = W1 x + b1 -> transposed [N x 32]; f2 -> packed-B; f3 -> row-major
    gemm_bf16_wmma<2><<<dim3(N / 512, C8 / 16, Bsz), blk, 0, stream>>>(
        w1b, xbp, (void*)f1t, b1, C8, N, C, (size_t)C * N, (size_t)C8 * N);
    gemm_bf16_wmma<1><<<dim3(N / 512, C8 / 16, Bsz), blk, 0, stream>>>(
        w2b, xbp, (void*)f2p, b2, C8, N, C, (size_t)C * N, (size_t)C8 * N);
    gemm_bf16_wmma<0><<<dim3(N / 512, C / 16, Bsz), blk, 0, stream>>>(
        w3b, xbp, (void*)f3b, b3, C, N, C, (size_t)C * N, (size_t)C * N);

    // Per batch: softmax(f1^T f2) -> shared attn (L2-resident), then
    // gated = (f3 @ attn) * x via LDS-staged GEMM. Stream order serializes.
    const size_t sh_bytes = (size_t)16 * N * 4 + 256 * 4 + 16 * 4;
    for (int b = 0; b < Bsz; ++b) {
        attn_logits_softmax<<<dim3(N / 16), blk, sh_bytes, stream>>>(
            f1t + (size_t)b * N * C8, f2p + (size_t)b * C8 * N, attn, N);
        attn_apply_wmma<<<dim3(N / BN, C / BM), blk, 0, stream>>>(
            f3b + (size_t)b * C * N, attn, x + (size_t)b * C * N,
            gbp + (size_t)b * C * N, C, N, N);
    }

    // out = Wo gated + bo (fp32)
    gemm_bf16_wmma<3><<<dim3(N / 512, C / 16, Bsz), blk, 0, stream>>>(
        wob, gbp, (void*)out, bo, C, N, C, (size_t)C * N, (size_t)C * N);
}
